// bd_attention_LSTM_4105988735700
// MI455X (gfx1250) — matrix operand order repaired and checked by reading, not compiled, not run on hardware
//
#include <hip/hip_runtime.h>
#include <hip/hip_bf16.h>

typedef __bf16 bf16;
typedef __attribute__((ext_vector_type(16))) __bf16 v16bf;
typedef __attribute__((ext_vector_type(8)))  float  v8f;
typedef __attribute__((ext_vector_type(4)))  unsigned int u32x4;
typedef __attribute__((ext_vector_type(8)))  unsigned int u32x8;

__device__ inline v8f wmma_bf16(v16bf a, v16bf b, v8f c) {
  // D = A(16x32 bf16) * B(32x16 bf16) + C(16x16 f32)
  return __builtin_amdgcn_wmma_f32_16x16x32_bf16(false, a, false, b, (short)0, c,
                                                 false, false);
}

__device__ inline float sigf(float x) { return 1.0f / (1.0f + __expf(-x)); }

// ---- fragment loaders (operands row-major with leading dim ld, lane-striped
//      per CDNA5 ISA 7.12.2: A lane -> row (l&15), K chunks {8h..},{16+8h..};
//      B lane -> col (l&15), same K chunks {8h..},{16+8h..}) ----
__device__ inline v16bf fragA(const bf16* base, int ld, int lane) {
  int r = lane & 15, h = lane >> 4;
  const bf16* p = base + (size_t)r * ld + 8 * h;
  union { v16bf v; u32x4 q[2]; } f;
  f.q[0] = *(const u32x4*)p;
  f.q[1] = *(const u32x4*)(p + 16);
  return f.v;
}
__device__ inline v16bf fragB(const bf16* base, int ld, int lane) {
  int n = lane & 15, h = lane >> 4;
  const bf16* p = base + (size_t)n * ld + 8 * h;
  union { v16bf v; u32x4 q[2]; } f;
  f.q[0] = *(const u32x4*)p;
  f.q[1] = *(const u32x4*)(p + 16);
  return f.v;
}
// same layouts but sourced from f32 (LDS staging filled by TDM, or global)
__device__ inline v16bf fragA32(const float* base, int ld, int lane) {
  int r = lane & 15, h = lane >> 4;
  const float* p = base + (size_t)r * ld + 8 * h;
  union { v16bf v; __bf16 e[16]; } f;
#pragma unroll
  for (int i = 0; i < 8; i++) {
    f.e[i]     = (__bf16)p[i];
    f.e[8 + i] = (__bf16)p[16 + i];
  }
  return f.v;
}
__device__ inline v16bf fragB32(const float* base, int ld, int lane) {
  int n = lane & 15, h = lane >> 4;
  const float* p = base + (size_t)n * ld + 8 * h;
  union { v16bf v; __bf16 e[16]; } f;
#pragma unroll
  for (int i = 0; i < 8; i++) {
    f.e[i]     = (__bf16)p[i];
    f.e[8 + i] = (__bf16)p[16 + i];
  }
  return f.v;
}
// A fragment for attention: (enc_row f32 + h_row bf16) -> bf16
__device__ inline v16bf fragA_att(const float* enc, const bf16* hrow, int k0, int lane) {
  int h = lane >> 4;
  union { v16bf v; __bf16 e[16]; } f;
#pragma unroll
  for (int i = 0; i < 8; i++) {
    int ka = k0 + 8 * h + i, kb = k0 + 16 + 8 * h + i;
    f.e[i]     = (__bf16)(enc[ka] + (float)hrow[ka]);
    f.e[8 + i] = (__bf16)(enc[kb] + (float)hrow[kb]);
  }
  return f.v;
}

// =====================================================================
// Tensor Data Mover: 2D tile (32 x rows, f32) global -> LDS, async.
// D# per CDNA5 ISA ch.8: group0 = {count=1 | lds_addr | global_addr | type=2},
// group1 = {data_size=4B, tensor_dim0=stride, tensor_dim1=rows,
//           tile_dim0=32, tile_dim1=rows, tensor_dim0_stride=stride}.
// Tracked by TENSORcnt; one wave issues, workgroup barrier publishes.
// =====================================================================
__device__ inline void tdm_load_tile_f32(const float* g, unsigned lds_off,
                                         unsigned stride_elems, unsigned rows) {
  unsigned long long ga = (unsigned long long)(size_t)g;
  u32x4 g0;
  g0[0] = 1u;                                              // count=1, user D#
  g0[1] = lds_off;                                         // lds_addr (bytes)
  g0[2] = (unsigned)ga;                                    // global_addr[31:0]
  g0[3] = ((unsigned)(ga >> 32) & 0x01FFFFFFu) | (2u << 30); // addr[56:32]|type=2
  u32x8 g1;
  g1[0] = (2u << 16);                                      // data_size = 4 bytes
  g1[1] = (stride_elems & 0xFFFFu) << 16;                  // tensor_dim0 lo16
  g1[2] = (stride_elems >> 16) | ((rows & 0xFFFFu) << 16); // dim0 hi | dim1 lo
  g1[3] = (32u << 16);                                     // dim1 hi=0 | tile_dim0=32
  g1[4] = rows & 0xFFFFu;                                  // tile_dim1 (tile_dim2=0)
  g1[5] = stride_elems;                                    // tensor_dim0_stride lo32
  g1[6] = 0u;                                              // stride hi | dim1_stride
  g1[7] = 0u;
  asm volatile("tensor_load_to_lds %0, %1" :: "s"(g0), "s"(g1) : "memory");
}

// =====================================================================
// Generic GEMM: C[M,N] = act(A[M,K](f32) @ W[N,K](f32)^T + bias[N])
// block = 256 threads (8 waves), 64x64 tile; grid (N/64, M/64).
// Double-buffered TDM pipeline: DMA stage ks+1 while WMMAs consume stage ks.
// =====================================================================
template <int ACT>
__global__ void gemm_wmma_kernel(const float* __restrict__ A, const float* __restrict__ W,
                                 const float* __restrict__ bias, float* __restrict__ C,
                                 int M, int N, int K) {
  __shared__ __align__(16) float Asf[2][64 * 32];
  __shared__ __align__(16) float Bsf[2][64 * 32];
  int tid = threadIdx.x;
  int wave = tid >> 5, lane = tid & 31;
  int lh = lane >> 4, lc = lane & 15;
  int m0 = blockIdx.y * 64, n0 = blockIdx.x * 64;
  int mt = wave & 3;
  int nt0 = (wave >> 2) * 2;
  const int nk = K >> 5;
  v8f c0 = {}, c1 = {};
  if (wave == 0) {  // prologue: stage ks=0 into buffer 0
    tdm_load_tile_f32(A + (size_t)m0 * K, (unsigned)(size_t)&Asf[0][0], (unsigned)K, 64u);
    tdm_load_tile_f32(W + (size_t)n0 * K, (unsigned)(size_t)&Bsf[0][0], (unsigned)K, 64u);
  }
  for (int ks = 0; ks < nk; ks++) {
    int cur = ks & 1;
    if (wave == 0) {
      if (ks + 1 < nk) {  // issue next stage, then wait for current pair only
        tdm_load_tile_f32(A + (size_t)m0 * K + (size_t)(ks + 1) * 32,
                          (unsigned)(size_t)&Asf[cur ^ 1][0], (unsigned)K, 64u);
        tdm_load_tile_f32(W + (size_t)n0 * K + (size_t)(ks + 1) * 32,
                          (unsigned)(size_t)&Bsf[cur ^ 1][0], (unsigned)K, 64u);
        __builtin_amdgcn_s_wait_tensorcnt(2);
      } else {
        __builtin_amdgcn_s_wait_tensorcnt(0);
      }
    }
    __syncthreads();  // publish staged tile to all waves
    v16bf af = fragA32(&Asf[cur][0] + mt * 16 * 32, 32, lane);
    v16bf b0 = fragB32(&Bsf[cur][0] + nt0 * 16 * 32, 32, lane);
    v16bf b1 = fragB32(&Bsf[cur][0] + (nt0 + 1) * 16 * 32, 32, lane);
    c0 = wmma_bf16(af, b0, c0);
    c1 = wmma_bf16(af, b1, c1);
    __syncthreads();  // all reads done before TDM overwrites this buffer
  }
#pragma unroll
  for (int j = 0; j < 8; j++) {
    int row = m0 + mt * 16 + 8 * lh + j;
    int cA = n0 + nt0 * 16 + lc;
    int cB = cA + 16;
    float vA = c0[j] + bias[cA];
    float vB = c1[j] + bias[cB];
    if (ACT) { vA = tanhf(vA); vB = tanhf(vB); }
    C[(size_t)row * N + cA] = vA;
    C[(size_t)row * N + cB] = vB;
  }
}

// =====================================================================
// LSTM recurrent scan. One workgroup, 512 threads (16 waves), persistent.
// z_t = Xz[t] + h_{t-1} @ Whh^T ; gates -> h,c.  Whh lives in LDS (bf16),
// c state lives in registers, h round-trips through a 16KB LDS bf16 buffer.
// =====================================================================
__global__ void lstm_scan_kernel(const float* __restrict__ Xz,   // [64*64, 512]
                                 const float* __restrict__ Whh,  // [512,128] f32
                                 float* __restrict__ out, int outLd, int outOff,
                                 int reverse, float* __restrict__ finalHC,
                                 int wantFinal) {
  extern __shared__ __align__(16) char smem[];
  bf16* whh  = (bf16*)smem;                    // 512*128 bf16 = 128KB
  bf16* hbuf = (bf16*)(smem + 512 * 128 * 2);  // 64*128  bf16 = 16KB
  int tid = threadIdx.x;
  int wave = tid >> 5, lane = tid & 31;
  int hj = wave & 7, mh = wave >> 3;
  int lh = lane >> 4, lc = lane & 15;
  for (int i = tid; i < 512 * 128; i += 512) whh[i] = (bf16)Whh[i];
  for (int i = tid; i < 64 * 128; i += 512) hbuf[i] = (bf16)0.0f;
  v8f cst[2] = {};
  __syncthreads();
  for (int t = 0; t < 64; t++) {
    int tt = reverse ? 63 - t : t;
    v8f acc[2][4] = {};
#pragma unroll
    for (int kk = 0; kk < 4; kk++) {
      v16bf a0 = fragA(hbuf + (size_t)(2 * mh + 0) * 16 * 128 + kk * 32, 128, lane);
      v16bf a1 = fragA(hbuf + (size_t)(2 * mh + 1) * 16 * 128 + kk * 32, 128, lane);
#pragma unroll
      for (int g = 0; g < 4; g++) {
        v16bf bf = fragB(whh + (size_t)(g * 128 + hj * 16) * 128 + kk * 32, 128, lane);
        acc[0][g] = wmma_bf16(a0, bf, acc[0][g]);
        acc[1][g] = wmma_bf16(a1, bf, acc[1][g]);
      }
    }
    __syncthreads();  // all reads of hbuf done
#pragma unroll
    for (int m = 0; m < 2; m++) {
      int mt = 2 * mh + m;
#pragma unroll
      for (int j = 0; j < 8; j++) {
        int b = mt * 16 + 8 * lh + j;
        int hid = hj * 16 + lc;
        const float* zrow = Xz + ((size_t)tt * 64 + b) * 512;
        float zi = acc[m][0][j] + zrow[hid];
        float zf = acc[m][1][j] + zrow[128 + hid];
        float zg = acc[m][2][j] + zrow[256 + hid];
        float zo = acc[m][3][j] + zrow[384 + hid];
        float cc = sigf(zf) * cst[m][j] + sigf(zi) * tanhf(zg);
        float hv = sigf(zo) * tanhf(cc);
        cst[m][j] = cc;
        hbuf[b * 128 + hid] = (bf16)hv;
        out[((size_t)tt * 64 + b) * outLd + outOff + hid] = hv;
        if (wantFinal && t == 63) {
          finalHC[b * 128 + hid] = hv;
          finalHC[8192 + b * 128 + hid] = cc;
        }
      }
    }
    __syncthreads();  // hbuf updated for next step
  }
}

// =====================================================================
// Decoder + attention, one persistent workgroup (512 threads / 16 waves).
// LDS: [Wih|Whh] combined 512x256 bf16 (256KB) -> one K=256 WMMA GEMM/step,
//      xh = [xin|h] 64x256 bf16, sc = scores 64x64 f32 (ds-atomic reduce).
// =====================================================================
__global__ void decoder_kernel(const float* __restrict__ enc_out,  // [64,64,128]
                               const float* __restrict__ ehc,      // [2][64][128]
                               const float* __restrict__ Wih,      // [512,128]
                               const float* __restrict__ Whh,      // [512,128]
                               const float* __restrict__ bvec,     // [512]
                               const float* __restrict__ attW,     // [128,128]
                               const float* __restrict__ attw,     // [128]
                               float* __restrict__ flat) {         // [64][16384]
  extern __shared__ __align__(16) char smem[];
  bf16* wc = (bf16*)smem;                                    // 512*256
  bf16* xh = (bf16*)(smem + 512 * 256 * 2);                  // 64*256
  float* sc = (float*)(smem + 512 * 256 * 2 + 64 * 256 * 2); // 64*64
  int tid = threadIdx.x, wave = tid >> 5, lane = tid & 31;
  int hj = wave & 7, mh = wave >> 3, lh = lane >> 4, lc = lane & 15;
  for (int i = tid; i < 512 * 256; i += 512) {
    int n = i >> 8, k = i & 255;
    wc[i] = (bf16)((k < 128) ? Wih[n * 128 + k] : Whh[n * 128 + k - 128]);
  }
  for (int i = tid; i < 64 * 128; i += 512) {
    int b = i >> 7, j = i & 127;
    xh[b * 256 + j]       = (bf16)enc_out[(size_t)(63 * 64 + b) * 128 + j];  // xin
    xh[b * 256 + 128 + j] = (bf16)ehc[i];                                     // h
  }
  v8f cst[2];
#pragma unroll
  for (int m = 0; m < 2; m++) {
    int mt = 2 * mh + m;
#pragma unroll
    for (int j = 0; j < 8; j++)
      cst[m][j] = ehc[8192 + (size_t)(mt * 16 + 8 * lh + j) * 128 + hj * 16 + lc];
  }
  __syncthreads();
  for (int t = 0; t < 64; t++) {
    // ---- 1. decoder cell: z = [xin|h] @ [Wih|Whh]^T + b ----
    v8f acc[2][4] = {};
#pragma unroll
    for (int kk = 0; kk < 8; kk++) {
      v16bf a0 = fragA(xh + (size_t)(2 * mh + 0) * 16 * 256 + kk * 32, 256, lane);
      v16bf a1 = fragA(xh + (size_t)(2 * mh + 1) * 16 * 256 + kk * 32, 256, lane);
#pragma unroll
      for (int g = 0; g < 4; g++) {
        v16bf bf = fragB(wc + (size_t)(g * 128 + hj * 16) * 256 + kk * 32, 256, lane);
        acc[0][g] = wmma_bf16(a0, bf, acc[0][g]);
        acc[1][g] = wmma_bf16(a1, bf, acc[1][g]);
      }
    }
    __syncthreads();
    // ---- 2. gates: register-resident c, write new h ----
#pragma unroll
    for (int m = 0; m < 2; m++) {
      int mt = 2 * mh + m;
#pragma unroll
      for (int j = 0; j < 8; j++) {
        int b = mt * 16 + 8 * lh + j;
        int hid = hj * 16 + lc;
        float zi = acc[m][0][j] + bvec[hid];
        float zf = acc[m][1][j] + bvec[128 + hid];
        float zg = acc[m][2][j] + bvec[256 + hid];
        float zo = acc[m][3][j] + bvec[384 + hid];
        float cc = sigf(zf) * cst[m][j] + sigf(zi) * tanhf(zg);
        float hv = sigf(zo) * tanhf(cc);
        cst[m][j] = cc;
        bf16 hb = (bf16)hv;
        xh[b * 256 + hid] = hb;        // xin_{t+1} = h_t
        xh[b * 256 + 128 + hid] = hb;  // h state
        flat[(size_t)b * 16384 + t * 256 + hid] = hv;
      }
    }
    for (int i = tid; i < 4096; i += 512) sc[i] = 0.0f;
    __syncthreads();
    // ---- 3. attention scores: tanh((h+enc) @ attW^T) @ attw ----
    for (int mt2 = wave; mt2 < 256; mt2 += 16) {
      int row = mt2 * 16 + lc;  // row = s*64+b, enc_out is linear in it
      const float* er = enc_out + (size_t)row * 128;
      const bf16* hr = xh + (row & 63) * 256 + 128;
      v16bf afr[4];
#pragma unroll
      for (int kk = 0; kk < 4; kk++) afr[kk] = fragA_att(er, hr, kk * 32, lane);
      float part[8] = {0, 0, 0, 0, 0, 0, 0, 0};
#pragma unroll
      for (int nt = 0; nt < 8; nt++) {
        v8f tac = {};
#pragma unroll
        for (int kk = 0; kk < 4; kk++) {
          v16bf bfr = fragB32(attW + (size_t)nt * 16 * 128 + kk * 32, 128, lane);
          tac = wmma_bf16(afr[kk], bfr, tac);
        }
        float wv = attw[nt * 16 + lc];
#pragma unroll
        for (int j = 0; j < 8; j++) part[j] += tanhf(tac[j]) * wv;
      }
#pragma unroll
      for (int j = 0; j < 8; j++) atomicAdd(&sc[mt2 * 16 + 8 * lh + j], part[j]);
    }
    __syncthreads();
    // ---- 4. softmax over s, per batch column ----
    if (tid < 64) {
      int b = tid;
      float mx = -3.4e38f;
      for (int s = 0; s < 64; s++) mx = fmaxf(mx, sc[s * 64 + b]);
      float sum = 0.0f;
      for (int s = 0; s < 64; s++) {
        float e = __expf(sc[s * 64 + b] - mx);
        sc[s * 64 + b] = e;
        sum += e;
      }
      float inv = 1.0f / sum;
      for (int s = 0; s < 64; s++) sc[s * 64 + b] *= inv;
    }
    __syncthreads();
    // ---- 5. context: ctx[b,j] = sum_s aw[s,b]*enc[s,b,j] ----
    for (int i = tid; i < 8192; i += 512) {
      int b = i >> 7, j = i & 127;
      float a2 = 0.0f;
      for (int s = 0; s < 64; s++)
        a2 += sc[s * 64 + b] * enc_out[((size_t)s * 64 + b) * 128 + j];
      flat[(size_t)b * 16384 + t * 256 + 128 + j] = a2;
    }
    __syncthreads();
  }
}

// final linear head: out[b] = x4[b,:] . out_W + out_b   (N=1, too small for WMMA)
__global__ void head_kernel(const float* __restrict__ x4, const float* __restrict__ outW,
                            const float* __restrict__ outb, float* __restrict__ out) {
  int b = threadIdx.x;  // 64 threads
  float a = 0.0f;
  for (int j = 0; j < 128; j++) a += x4[b * 128 + j] * outW[j];
  out[b] = a + outb[0];
}

extern "C" void kernel_launch(void* const* d_in, const int* in_sizes, int n_in,
                              void* d_out, int out_size, void* d_ws, size_t ws_size,
                              hipStream_t stream) {
  (void)in_sizes; (void)n_in; (void)out_size; (void)ws_size;
  const float* input = (const float*)d_in[0];
  const float* l1_Wih = (const float*)d_in[1], *l1_Whh = (const float*)d_in[2], *l1_b = (const float*)d_in[3];
  const float* f2_Wih = (const float*)d_in[4], *f2_Whh = (const float*)d_in[5], *f2_b = (const float*)d_in[6];
  const float* b2_Wih = (const float*)d_in[7], *b2_Whh = (const float*)d_in[8], *b2_b = (const float*)d_in[9];
  const float* f3_Wih = (const float*)d_in[10], *f3_Whh = (const float*)d_in[11], *f3_b = (const float*)d_in[12];
  const float* b3_Wih = (const float*)d_in[13], *b3_Whh = (const float*)d_in[14], *b3_b = (const float*)d_in[15];
  const float* en_Wih = (const float*)d_in[16], *en_Whh = (const float*)d_in[17], *en_b = (const float*)d_in[18];
  const float* de_Wih = (const float*)d_in[19], *de_Whh = (const float*)d_in[20], *de_b = (const float*)d_in[21];
  const float* attW = (const float*)d_in[22], *attw = (const float*)d_in[23];
  const float* W1 = (const float*)d_in[24], *W1b = (const float*)d_in[25];
  const float* W2 = (const float*)d_in[26], *W2b = (const float*)d_in[27];
  const float* W3 = (const float*)d_in[28], *W3b = (const float*)d_in[29];
  const float* W4 = (const float*)d_in[30], *W4b = (const float*)d_in[31];
  const float* oW = (const float*)d_in[32], *ob = (const float*)d_in[33];
  float* out = (float*)d_out;

  char* p = (char*)d_ws;
  auto alloc = [&](size_t bytes) -> void* {
    void* r = (void*)p;
    p += (bytes + 255) & ~(size_t)255;
    return r;
  };
  float* Xz   = (float*)alloc((size_t)4096 * 512 * 4);
  float* x1   = (float*)alloc((size_t)4096 * 128 * 4);
  float* x2   = (float*)alloc((size_t)4096 * 256 * 4);
  float* x3   = (float*)alloc((size_t)4096 * 256 * 4);
  float* enc  = (float*)alloc((size_t)4096 * 128 * 4);
  float* ehc  = (float*)alloc((size_t)2 * 64 * 128 * 4);
  float* flat = (float*)alloc((size_t)64 * 16384 * 4);
  float* m1   = (float*)alloc((size_t)64 * 1024 * 4);
  float* m2   = (float*)alloc((size_t)64 * 512 * 4);
  float* m3   = (float*)alloc((size_t)64 * 256 * 4);
  float* m4   = (float*)alloc((size_t)64 * 128 * 4);

  const size_t scanLds = (size_t)512 * 128 * 2 + (size_t)64 * 128 * 2;             // 144KB
  const size_t decLds  = (size_t)512 * 256 * 2 + (size_t)64 * 256 * 2 + 4096 * 4;  // 304KB
  dim3 blk(256);

  // ---- layer 1 (unidirectional) ----
  gemm_wmma_kernel<0><<<dim3(8, 64), blk, 0, stream>>>(input, l1_Wih, l1_b, Xz, 4096, 512, 64);
  lstm_scan_kernel<<<1, 512, scanLds, stream>>>(Xz, l1_Whh, x1, 128, 0, 0, nullptr, 0);
  // ---- bi-LSTM 2 ----
  gemm_wmma_kernel<0><<<dim3(8, 64), blk, 0, stream>>>(x1, f2_Wih, f2_b, Xz, 4096, 512, 128);
  lstm_scan_kernel<<<1, 512, scanLds, stream>>>(Xz, f2_Whh, x2, 256, 0, 0, nullptr, 0);
  gemm_wmma_kernel<0><<<dim3(8, 64), blk, 0, stream>>>(x1, b2_Wih, b2_b, Xz, 4096, 512, 128);
  lstm_scan_kernel<<<1, 512, scanLds, stream>>>(Xz, b2_Whh, x2, 256, 128, 1, nullptr, 0);
  // ---- bi-LSTM 3 ----
  gemm_wmma_kernel<0><<<dim3(8, 64), blk, 0, stream>>>(x2, f3_Wih, f3_b, Xz, 4096, 512, 256);
  lstm_scan_kernel<<<1, 512, scanLds, stream>>>(Xz, f3_Whh, x3, 256, 0, 0, nullptr, 0);
  gemm_wmma_kernel<0><<<dim3(8, 64), blk, 0, stream>>>(x2, b3_Wih, b3_b, Xz, 4096, 512, 256);
  lstm_scan_kernel<<<1, 512, scanLds, stream>>>(Xz, b3_Whh, x3, 256, 128, 1, nullptr, 0);
  // ---- encoder ----
  gemm_wmma_kernel<0><<<dim3(8, 64), blk, 0, stream>>>(x3, en_Wih, en_b, Xz, 4096, 512, 256);
  lstm_scan_kernel<<<1, 512, scanLds, stream>>>(Xz, en_Whh, enc, 128, 0, 0, ehc, 1);
  // ---- decoder + attention ----
  decoder_kernel<<<1, 512, decLds, stream>>>(enc, ehc, de_Wih, de_Whh, de_b, attW, attw, flat);
  // ---- MLP ----
  gemm_wmma_kernel<1><<<dim3(16, 1), blk, 0, stream>>>(flat, W1, W1b, m1, 64, 1024, 16384);
  gemm_wmma_kernel<1><<<dim3(8, 1),  blk, 0, stream>>>(m1, W2, W2b, m2, 64, 512, 1024);
  gemm_wmma_kernel<1><<<dim3(4, 1),  blk, 0, stream>>>(m2, W3, W3b, m3, 64, 256, 512);
  gemm_wmma_kernel<1><<<dim3(2, 1),  blk, 0, stream>>>(m3, W4, W4b, m4, 64, 128, 256);
  head_kernel<<<1, 64, 0, stream>>>(m4, oW, ob, out);
}
